// Net_70781061038143
// MI455X (gfx1250) — compile-verified, matrix operand order checked by reading, not run on hardware
//
#include <hip/hip_runtime.h>
#include <hip/hip_bf16.h>
#include <math.h>

typedef __attribute__((ext_vector_type(16))) _Float16 v16h;
typedef __attribute__((ext_vector_type(8)))  float    v8f;

#define NN 50000
#define NE 800000
#define NG 64

// ---------------------------------------------------------------------------
// degree / norm
// ---------------------------------------------------------------------------
__global__ void deg_init_kernel(float* deg, int n) {
    int i = blockIdx.x * blockDim.x + threadIdx.x;
    if (i < n) deg[i] = 1.0f;            // self-loop contributes 1 to every node
}

__global__ void deg_accum_kernel(const int* __restrict__ dst, float* deg, int E) {
    int i = blockIdx.x * blockDim.x + threadIdx.x;
    if (i < E) atomicAdd(&deg[dst[i]], 1.0f);
}

__global__ void norm_kernel(const float* __restrict__ deg,
                            const int* __restrict__ src, const int* __restrict__ dst,
                            float* __restrict__ normv, int E, int Etot) {
    int i = blockIdx.x * blockDim.x + threadIdx.x;
    if (i >= Etot) return;
    int s, d;
    if (i < E) { s = src[i]; d = dst[i]; } else { s = d = i - E; }
    normv[i] = rsqrtf(deg[s]) * rsqrtf(deg[d]);   // deg >= 1 always
}

__global__ void zero_kernel(float* p, long long n) {
    long long i = (long long)blockIdx.x * blockDim.x + threadIdx.x;
    if (i < n) p[i] = 0.0f;
}

// ---------------------------------------------------------------------------
// f32 -> f16 conversions (once per GEMM operand, hoisted out of the hot loop)
// ---------------------------------------------------------------------------
__global__ void cvt_kernel(const float* __restrict__ src, _Float16* __restrict__ dst, long long n) {
    long long i = (long long)blockIdx.x * blockDim.x + threadIdx.x;
    if (i < n) dst[i] = (_Float16)src[i];
}

// W [K x M] f32 row-major  ->  Wt [M x K] f16 row-major (transpose + convert)
__global__ void cvtT_kernel(const float* __restrict__ W, _Float16* __restrict__ Wt,
                            int K, int M, int mshift) {
    int i = blockIdx.x * blockDim.x + threadIdx.x;
    if (i >= K * M) return;
    int k = i >> mshift;
    int mcol = i & (M - 1);
    Wt[mcol * K + k] = (_Float16)W[i];
}

// ---------------------------------------------------------------------------
// WMMA GEMM: C[NN x M] = A[NN x K]h @ B[M x K]h^T (+bias1+bias2), f32 accum
// A, B already f16.  B is [M x K] row-major (K contiguous -> dword pair loads).
// Block = 128 threads (4 waves); each wave owns TWO 16x16 C tiles (32 cols).
// A tile (16x32 f16 = 1KB) double-buffered in LDS; filled one K-step ahead by
// global_load_async_to_lds_b64 so the DMA overlaps the WMMAs (ASYNCcnt).
// K is < 32 (single padded step) or a multiple of 32 (guard-free path).
// ---------------------------------------------------------------------------
__device__ __forceinline__ v8f wmma_f16(const v16h& a, const v16h& b, const v8f& c) {
    return __builtin_amdgcn_wmma_f32_16x16x32_f16(false, a, false, b, (short)0, c, false, false);
}

__global__ void gemm_wmma_kernel(const _Float16* __restrict__ A, int lda,
                                 const _Float16* __restrict__ B, int K, int M,
                                 const float* __restrict__ bias1, const float* __restrict__ bias2,
                                 float* __restrict__ C, int ldc) {
    __shared__ _Float16 lds_a[2][16 * 32];               // 2 x 1KB ping-pong

    const int lane  = threadIdx.x & 31;
    const int wave  = threadIdx.x >> 5;                  // 0..3
    const int rowTile = blockIdx.y * 16;
    const int col0  = blockIdx.x * 128 + wave * 32;      // tiles at col0 and col0+16
    const int m     = lane & 15;                         // A-row / B-col / C-col
    const int half  = lane >> 4;
    const bool act0 = (col0 < M);                        // wave-uniform
    const bool act1 = (col0 + 16 < M);
    const int n0 = col0 + m, n1 = col0 + 16 + m;

    union Frag { v16h h; unsigned u[8]; };
    Frag av, bv0, bv1;
    v8f acc0 = {}, acc1 = {};

    const _Float16* b0p = B + n0 * K;                    // valid only if act0
    const _Float16* b1p = B + n1 * K;                    // valid only if act1

    if (K >= 32) {                                       // K % 32 == 0 by construction
        const int r = threadIdx.x >> 3;                  // staging row 0..15
        const int c = (threadIdx.x & 7) * 4;             // staging col (halves)
        const unsigned lbase = (unsigned)(size_t)&lds_a[0][0];
        const unsigned soff  = (unsigned)(r * 32 + c) * 2u;
        const _Float16* ap = A + (rowTile + r) * lda + c;

        // prologue: fill buffer 0 with the first tile
#if defined(__gfx1250__)
        asm volatile("global_load_async_to_lds_b64 %0, %1, off"
                     :: "v"(lbase + soff), "v"((unsigned long long)(size_t)ap)
                     : "memory");
#else
        for (int t = 0; t < 4; ++t) lds_a[0][r * 32 + c + t] = ap[t];
#endif
        int cur = 0;
        for (int k0 = 0; k0 < K; k0 += 32, cur ^= 1) {
#if defined(__gfx1250__)
            asm volatile("s_wait_asynccnt 0" ::: "memory");   // tile `cur` landed
#endif
            __syncthreads();

            if (k0 + 32 < K) {                           // start DMA of next tile now
#if defined(__gfx1250__)
                asm volatile("global_load_async_to_lds_b64 %0, %1, off"
                             :: "v"(lbase + (unsigned)(cur ^ 1) * 1024u + soff),
                                "v"((unsigned long long)(size_t)(ap + k0 + 32))
                             : "memory");
#else
                for (int t = 0; t < 4; ++t) lds_a[cur ^ 1][r * 32 + c + t] = ap[k0 + 32 + t];
#endif
                if (act0) __builtin_prefetch(b0p + k0 + 32, 0, 1);
                if (act1) __builtin_prefetch(b1p + k0 + 32, 0, 1);
            }

            const unsigned* lds_u = (const unsigned*)&lds_a[cur][0];
#pragma unroll
            for (int vv = 0; vv < 8; ++vv) {
                const int kk = ((vv >> 2) * 16) + half * 8 + ((vv & 3) * 2);
                av.u[vv] = lds_u[(m * 32 + kk) >> 1];
                unsigned u0 = 0u, u1 = 0u;
                if (act0) u0 = *(const unsigned*)(b0p + k0 + kk);
                if (act1) u1 = *(const unsigned*)(b1p + k0 + kk);
                bv0.u[vv] = u0;
                bv1.u[vv] = u1;
            }
            acc0 = wmma_f16(av.h, bv0.h, acc0);
            acc1 = wmma_f16(av.h, bv1.h, acc1);
            __syncthreads();                             // done reading buf `cur`
        }
    } else {
        // single partial K step (K == 1 or 16), zero-padded
        for (int t = threadIdx.x; t < 512; t += blockDim.x) {
            int r = t >> 5, kk = t & 31;
            lds_a[0][r * 32 + kk] = (kk < K) ? A[(rowTile + r) * lda + kk] : (_Float16)0.0f;
        }
        __syncthreads();

        const unsigned* lds_u = (const unsigned*)&lds_a[0][0];
#pragma unroll
        for (int vv = 0; vv < 8; ++vv) {
            const int kk = ((vv >> 2) * 16) + half * 8 + ((vv & 3) * 2);
            av.u[vv] = lds_u[(m * 32 + kk) >> 1];
            _Float16 z0 = (_Float16)0.0f;
            _Float16 e00 = z0, e01 = z0, e10 = z0, e11 = z0;
            if (act0) { if (kk < K) e00 = b0p[kk]; if (kk + 1 < K) e01 = b0p[kk + 1]; }
            if (act1) { if (kk < K) e10 = b1p[kk]; if (kk + 1 < K) e11 = b1p[kk + 1]; }
            bv0.h[2 * vv] = e00; bv0.h[2 * vv + 1] = e01;
            bv1.h[2 * vv] = e10; bv1.h[2 * vv + 1] = e11;
        }
        acc0 = wmma_f16(av.h, bv0.h, acc0);
        acc1 = wmma_f16(av.h, bv1.h, acc1);
    }

    // epilogue (32-bit indexing; max index ~51M)
    if (act0) {
        float bs = 0.f;
        if (bias1) bs += bias1[n0];
        if (bias2) bs += bias2[n0];
#pragma unroll
        for (int vv = 0; vv < 8; ++vv)
            C[(rowTile + vv + 8 * half) * ldc + n0] = acc0[vv] + bs;
    }
    if (act1) {
        float bs = 0.f;
        if (bias1) bs += bias1[n1];
        if (bias2) bs += bias2[n1];
#pragma unroll
        for (int vv = 0; vv < 8; ++vv)
            C[(rowTile + vv + 8 * half) * ldc + n1] = acc1[vv] + bs;
    }
}

// ---------------------------------------------------------------------------
// edge scatter-sum: agg[dst] += z[src] * norm (float4 per thread; self loops at e>=E)
// qshift = log2(F/4)
// ---------------------------------------------------------------------------
__global__ void scatter_kernel(const float* __restrict__ z, const float* __restrict__ normv,
                               const int* __restrict__ src, const int* __restrict__ dst,
                               int E, int Etot, int F, int qshift, float* __restrict__ agg) {
    long long tid = (long long)blockIdx.x * blockDim.x + threadIdx.x;
    long long total = (long long)Etot << qshift;
    if (tid >= total) return;
    int e = (int)(tid >> qshift);
    int f = (int)(tid & ((1 << qshift) - 1)) * 4;
    int s, d;
    if (e < E) { s = src[e]; d = dst[e]; } else { s = d = e - E; }
    float nv = normv[e];
    const float4 zv = *(const float4*)&z[(long long)s * F + f];
    float* ap = &agg[(long long)d * F + f];
    atomicAdd(ap + 0, zv.x * nv);
    atomicAdd(ap + 1, zv.y * nv);
    atomicAdd(ap + 2, zv.z * nv);
    atomicAdd(ap + 3, zv.w * nv);
}

__global__ void bias_relu_kernel(const float* __restrict__ agg, const float* __restrict__ b,
                                 float* __restrict__ h, long long total, int fmask) {
    long long i = (long long)blockIdx.x * blockDim.x + threadIdx.x;
    if (i >= total) return;
    float v = agg[i] + b[(int)(i & fmask)];
    h[i] = v > 0.0f ? v : 0.0f;
}

// ---------------------------------------------------------------------------
// LSTM cell activations (seq_len 1, zero state): c = sig(i)*tanh(g); h = sig(o)*tanh(c)
// ---------------------------------------------------------------------------
__device__ __forceinline__ float sigmoidf_(float x) { return 1.0f / (1.0f + expf(-x)); }

__global__ void lstm_act_kernel(const float* __restrict__ gates, float* __restrict__ hout,
                                long long total, int H, int hshift) {
    long long i = (long long)blockIdx.x * blockDim.x + threadIdx.x;
    if (i >= total) return;
    long long n = i >> hshift;
    int f = (int)(i & (H - 1));
    const float* g = gates + n * (long long)(4 * H);
    float ig = g[f];
    float gg = g[2 * H + f];
    float og = g[3 * H + f];
    float c  = sigmoidf_(ig) * tanhf(gg);
    hout[i]  = sigmoidf_(og) * tanhf(c);
}

__global__ void add_kernel(float* __restrict__ a, const float* __restrict__ bsrc, long long n) {
    long long i = (long long)blockIdx.x * blockDim.x + threadIdx.x;
    if (i < n) a[i] += bsrc[i];
}

// ---------------------------------------------------------------------------
// segment-max pooling (post-ReLU values >= 0, init 0 matches isfinite guard)
// ---------------------------------------------------------------------------
__global__ void pool_kernel(const float* __restrict__ h, const int* __restrict__ batch,
                            float* __restrict__ pooled, int n, int F) {
    long long i = (long long)blockIdx.x * blockDim.x + threadIdx.x;
    long long total = (long long)n * F;
    if (i >= total) return;
    int node = (int)(i / F);
    int f    = (int)(i - (long long)node * F);
    int g    = batch[node];
    atomicMax((int*)&pooled[(long long)g * F + f], __float_as_int(h[i]));
}

__global__ void cls_kernel(const float* __restrict__ pooled, const float* __restrict__ W,
                           const float* __restrict__ b, float* __restrict__ out) {
    int t = blockIdx.x * blockDim.x + threadIdx.x;   // 128 outputs
    if (t >= NG * 2) return;
    int g = t >> 1, j = t & 1;
    float s = b[j];
    for (int f = 0; f < 512; ++f) s += pooled[g * 512 + f] * W[f * 2 + j];
    out[t] = s;
}

// ---------------------------------------------------------------------------
// host orchestration
// ---------------------------------------------------------------------------
static inline void launch_gemm(const _Float16* A, int lda, const _Float16* B, int K, int M,
                               const float* b1, const float* b2, float* C, int ldc,
                               hipStream_t s) {
    dim3 grid((M + 127) / 128, NN / 16);
    gemm_wmma_kernel<<<grid, 128, 0, s>>>(A, lda, B, K, M, b1, b2, C, ldc);
}

static inline int ilog2(int x) { int r = 0; while ((1 << r) < x) ++r; return r; }

extern "C" void kernel_launch(void* const* d_in, const int* in_sizes, int n_in,
                              void* d_out, int out_size, void* d_ws, size_t ws_size,
                              hipStream_t stream) {
    (void)in_sizes; (void)n_in; (void)out_size; (void)ws_size;
    const int DIMS[7] = {1, 16, 32, 64, 128, 256, 512};

    const float* x    = (const float*)d_in[0];
    const int*   ei   = (const int*)d_in[1];      // [2, E]
    const int*   bat  = (const int*)d_in[2];
    const int*   esrc = ei;
    const int*   edst = ei + NE;
    const float* gW[6]; const float* gB[6];
    for (int i = 0; i < 6; ++i) { gW[i] = (const float*)d_in[3 + 2 * i]; gB[i] = (const float*)d_in[4 + 2 * i]; }
    // lstm params: 5 stacks x 4 layers x (Wih, Whh, bih, bhh) starting at index 15
    const float* cW = (const float*)d_in[95];
    const float* cB = (const float*)d_in[96];
    float* out = (float*)d_out;

    const long long N = NN;
    const int E = NE, Etot = NE + NN;

    float* ws     = (float*)d_ws;
    float* h      = ws;                    // N*512
    float* z      = ws + N * 512;          // N*512
    float* agg    = ws + N * 1024;         // N*512
    float* gts    = ws + N * 512;          // N*1024 (aliases z+agg; LSTM phase only)
    float* hA     = ws + N * 1536;         // N*256
    float* hB     = ws + N * 1792;         // N*256
    float* normv  = ws + N * 2048;         // Etot
    float* deg    = normv + Etot;          // N
    float* pooled = deg + N;               // 64*512
    _Float16* ah  = (_Float16*)(pooled + NG * 512);   // N*512 halves (8B aligned)
    _Float16* bh  = ah + N * 512;                     // up to 1024*512 halves

    // --- GCN normalization coefficients ---
    deg_init_kernel<<<(NN + 255) / 256, 256, 0, stream>>>(deg, NN);
    deg_accum_kernel<<<(E + 255) / 256, 256, 0, stream>>>(edst, deg, E);
    norm_kernel<<<(Etot + 255) / 256, 256, 0, stream>>>(deg, esrc, edst, normv, E, Etot);

    // --- GCN layer 0: h = relu(scatter(x @ W0) + b0) ---
    {
        int F = DIMS[1];
        cvt_kernel<<<(unsigned)((N + 255) / 256), 256, 0, stream>>>(x, ah, N);
        cvtT_kernel<<<1, 64, 0, stream>>>(gW[0], bh, 1, F, ilog2(F));
        launch_gemm(ah, 1, bh, 1, F, nullptr, nullptr, z, F, stream);
        zero_kernel<<<(unsigned)((N * F + 255) / 256), 256, 0, stream>>>(agg, N * F);
        int qs = ilog2(F) - 2;
        long long tot = (long long)Etot << qs;
        scatter_kernel<<<(unsigned)((tot + 255) / 256), 256, 0, stream>>>(z, normv, esrc, edst, E, Etot, F, qs, agg);
        bias_relu_kernel<<<(unsigned)((N * F + 255) / 256), 256, 0, stream>>>(agg, gB[0], h, N * F, F - 1);
    }

    // --- 5 x (residual LSTM stack + GCN layer) ---
    for (int i = 0; i < 5; ++i) {
        int H = DIMS[i + 1];
        int hsh = ilog2(H);
        const float* cur = h;                 // input dim == H
        for (int l = 0; l < 4; ++l) {
            int pi = 15 + i * 16 + l * 4;
            const float* Wih = (const float*)d_in[pi + 0];   // [4H x H] row-major == B[M x K]
            const float* bih = (const float*)d_in[pi + 2];
            const float* bhh = (const float*)d_in[pi + 3];
            cvt_kernel<<<(unsigned)((N * H + 255) / 256), 256, 0, stream>>>(cur, ah, N * H);
            cvt_kernel<<<(4 * H * H + 255) / 256, 256, 0, stream>>>(Wih, bh, (long long)4 * H * H);
            launch_gemm(ah, H, bh, H, 4 * H, bih, bhh, gts, 4 * H, stream);
            float* nxt = (l & 1) ? hB : hA;
            lstm_act_kernel<<<(unsigned)((N * H + 255) / 256), 256, 0, stream>>>(gts, nxt, N * H, H, hsh);
            cur = nxt;
        }
        add_kernel<<<(unsigned)((N * H + 255) / 256), 256, 0, stream>>>(h, cur, N * H);

        int F = DIMS[i + 2];
        cvt_kernel<<<(unsigned)((N * H + 255) / 256), 256, 0, stream>>>(h, ah, N * H);
        cvtT_kernel<<<(H * F + 255) / 256, 256, 0, stream>>>(gW[i + 1], bh, H, F, ilog2(F));
        launch_gemm(ah, H, bh, H, F, nullptr, nullptr, z, F, stream);
        zero_kernel<<<(unsigned)((N * F + 255) / 256), 256, 0, stream>>>(agg, N * F);
        int qs = ilog2(F) - 2;
        long long tot = (long long)Etot << qs;
        scatter_kernel<<<(unsigned)((tot + 255) / 256), 256, 0, stream>>>(z, normv, esrc, edst, E, Etot, F, qs, agg);
        bias_relu_kernel<<<(unsigned)((N * F + 255) / 256), 256, 0, stream>>>(agg, gB[i + 1], h, N * F, F - 1);
    }

    // --- pooling + classifier ---
    zero_kernel<<<(NG * 512 + 255) / 256, 256, 0, stream>>>(pooled, NG * 512);
    pool_kernel<<<(unsigned)((N * 512 + 255) / 256), 256, 0, stream>>>(h, bat, pooled, NN, 512);
    cls_kernel<<<1, 128, 0, stream>>>(pooled, cW, cB, out);
}
